// RNNPool_5583457485584
// MI455X (gfx1250) — matrix operand order repaired and checked by reading, not compiled, not run on hardware
//
#include <hip/hip_runtime.h>

typedef __bf16 bf16;
typedef __attribute__((ext_vector_type(16))) __bf16 v16bf;
typedef __attribute__((ext_vector_type(8)))  __bf16 v8bf;
typedef __attribute__((ext_vector_type(8)))  float  v8f;
typedef __attribute__((ext_vector_type(4)))  unsigned int u32x4;
typedef __attribute__((ext_vector_type(4)))  float  f32x4;

// explicit LDS (addrspace 3) types: guarantees ds_load/ds_store, 32-bit addressing
typedef __attribute__((address_space(3))) bf16        lds_bf16;
typedef __attribute__((address_space(3))) const u32x4 lds_cu32x4;

#define DDIM   1024
#define KSTEPS 8
#define NROWS  4096          // B * n_windows = 8 * 512
#define MTILE  32
#define LDP    1032          // padded LDS row (bf16 elems): stride 2064B -> conflict-free
#define BUFE   (MTILE * LDP)               // elements per LDS buffer
#define SMEM_BYTES (4 * BUFE * 2)          // xs ping/pong + h ping/pong = 264192 B
#define FRAG_ELEMS 512       // one packed 32K x 16N bf16 fragment = 1 KB
#define NTILE_STRIDE (32 * FRAG_ELEMS)     // 32 K-tiles per n-tile

static __device__ __forceinline__ bf16 f2bf(float f) {
    unsigned u = __builtin_bit_cast(unsigned, f);
    unsigned r = u + 0x7fffu + ((u >> 16) & 1u);     // round-to-nearest-even
    unsigned short h = (unsigned short)(r >> 16);
    return __builtin_bit_cast(bf16, h);
}

// Branchless tanh: native v_tanh_f32 (confirmed present on gfx1250).
static __device__ __forceinline__ float tanh_fast(float x) {
#if __has_builtin(__builtin_amdgcn_tanhf)
    return __builtin_amdgcn_tanhf(x);
#elif __has_builtin(__builtin_amdgcn_tanh_f32)
    return __builtin_amdgcn_tanh_f32(x);
#else
    const float e = __builtin_amdgcn_exp2f(x * 2.8853900817779268f);
    const float r = __builtin_amdgcn_rcpf(e + 1.0f);
    return __builtin_fmaf(-2.0f, r, 1.0f);
#endif
}

union Frag { v16bf f; u32x4 q[2]; };

// ---- pack W_ih/W_hh into WMMA-B-fragment-contiguous bf16 layout ----
// Fragment (ntile, kt) covers n = ntile*16 + l16, k = kt*32 + half*8 .. +7 and kt*32 + 16 + half*8 .. +7.
// Lane L's 16 elements live at fragbase + L*16 -> two coalesced b128 per fragment.
__global__ void pack_kernel(const float* __restrict__ Wih, const float* __restrict__ Whh,
                            const float* __restrict__ bih, const float* __restrict__ bhh,
                            bf16* __restrict__ wpi, bf16* __restrict__ wph,
                            float* __restrict__ bias) {
    const int t = blockIdx.x * blockDim.x + threadIdx.x;   // 65536 threads
    const int lane  = t & 31;
    const int kt    = (t >> 5) & 31;
    const int ntile = t >> 10;                              // 0..63
    const int n  = ntile * 16 + (lane & 15);
    const int kb = kt * 32 + (lane >> 4) * 8;
    const size_t src = (size_t)n * DDIM + kb;
    const size_t dst = ((size_t)ntile * 32 + kt) * FRAG_ELEMS + lane * 16;
#pragma unroll
    for (int j = 0; j < 16; ++j) {
        wpi[dst + j] = f2bf(Wih[src + j + (j & 8)]);
        wph[dst + j] = f2bf(Whh[src + j + (j & 8)]);
    }
    if (t < DDIM) bias[t] = bih[t] + bhh[t];
}

// ---- convert x to bf16 once (feeds async LDS staging in the main kernel) ----
__global__ void xconv_kernel(const float* __restrict__ x, bf16* __restrict__ xbf) {
    const size_t i = ((size_t)blockIdx.x * blockDim.x + threadIdx.x) * 8;
    const f32x4 a = *(const f32x4*)(x + i);
    const f32x4 b = *(const f32x4*)(x + i + 4);
    v8bf o = { f2bf(a.x), f2bf(a.y), f2bf(a.z), f2bf(a.w),
               f2bf(b.x), f2bf(b.y), f2bf(b.z), f2bf(b.w) };
    *(v8bf*)(xbf + i) = o;
}

__global__ __launch_bounds__(256)
void rnnpool_kernel(const bf16* __restrict__ xbf,
                    const bf16* __restrict__ wpi, const bf16* __restrict__ wph,
                    const float* __restrict__ bias, float* __restrict__ out) {
    extern __shared__ __align__(16) char smem[];
    lds_bf16* const lds = (lds_bf16*)smem;
    lds_bf16* const xs0 = lds;                 // staged x_k ping
    lds_bf16* const xs1 = lds + BUFE;          // staged x_k pong
    lds_bf16* const hb0 = lds + 2 * BUFE;      // hidden state ping
    lds_bf16* const hb1 = lds + 3 * BUFE;      // hidden state pong

    const int tid  = threadIdx.x;
    const int wave = tid >> 5;
    const int lane = tid & 31;
    const int half = lane >> 4;
    const int l16  = lane & 15;
    const int r0   = blockIdx.x * MTILE;

    // h0 = 0
    for (int i = tid; i < BUFE; i += 256) hb0[i] = f2bf(0.0f);

    // A-fragment addressing (16-bit A 16x32): lane l16 = row M,
    // VGPR0-3 = K[half*8..+7], VGPR4-7 = K[16+half*8..+7]
    const int aRow0 = l16 * LDP;                 // tile-row 0
    const int aRow1 = (16 + l16) * LDP;          // tile-row 1
    const int aK0   = half * 8;

    // ---- async x staging: 32 rows x 2048 B as 16 x b128 per thread ----
    auto issue_stage = [&](int k) {
        // AS3 pointer value == wave-relative LDS byte offset (async VDST semantics)
        const unsigned base = (unsigned)(__SIZE_TYPE__)((k & 1) ? xs1 : xs0);
#pragma unroll
        for (int t = 0; t < 16; ++t) {
            const int q   = tid + t * 256;
            const int row = q >> 7;              // 0..31
            const int seg = q & 127;             // 16-byte segment within row
            const bf16* g = xbf + ((size_t)(r0 + row) * KSTEPS + k) * DDIM + seg * 8;
            const unsigned l = base + row * (LDP * 2) + seg * 16;
            asm volatile("global_load_async_to_lds_b128 %0, %1, off"
                         :: "v"(l), "v"(g) : "memory");
        }
    };

    issue_stage(0);   // prefetch step 0 while h0 init settles

    for (int k = 0; k < KSTEPS; ++k) {
        lds_bf16* const cur = (k & 1) ? hb1 : hb0;
        lds_bf16* const nxt = (k & 1) ? hb0 : hb1;
        const lds_bf16* const xs = (k & 1) ? xs1 : xs0;

        // wait for this wave's async copies of xs[k&1]; barrier orders everyone's
        // copies AND the previous step's h writes / xs reads.
        asm volatile("s_wait_asynccnt 0x0" ::: "memory");
        __syncthreads();

        if (k + 1 < KSTEPS) issue_stage(k + 1);   // overlap next stage with compute

        // ---- two 512-col chunks; each wave: 2 tile-rows x 4 tile-cols ----
        for (int nc = 0; nc < 2; ++nc) {
            const int nb = nc * 512 + wave * 64; // this wave's first column
            v8f acc[2][4];
            const bf16* pi[4];
            const bf16* ph[4];
#pragma unroll
            for (int c = 0; c < 4; ++c) {
                const int col   = nb + c * 16 + l16;
                const int ntile = (nb >> 4) + c;
                const float bv  = bias[col];
#pragma unroll
                for (int v = 0; v < 8; ++v) { acc[0][c][v] = bv; acc[1][c][v] = bv; }
                pi[c] = wpi + (size_t)ntile * NTILE_STRIDE + lane * 16;
                ph[c] = wph + (size_t)ntile * NTILE_STRIDE + lane * 16;
            }
#pragma unroll 1
            for (int kk = 0; kk < DDIM; kk += 32) {
                Frag ax0, ax1, ah0, ah1;
                ax0.q[0] = *(lds_cu32x4*)(xs  + aRow0 + kk + aK0);
                ax0.q[1] = *(lds_cu32x4*)(xs  + aRow0 + kk + aK0 + 16);
                ax1.q[0] = *(lds_cu32x4*)(xs  + aRow1 + kk + aK0);
                ax1.q[1] = *(lds_cu32x4*)(xs  + aRow1 + kk + aK0 + 16);
                ah0.q[0] = *(lds_cu32x4*)(cur + aRow0 + kk + aK0);
                ah0.q[1] = *(lds_cu32x4*)(cur + aRow0 + kk + aK0 + 16);
                ah1.q[0] = *(lds_cu32x4*)(cur + aRow1 + kk + aK0);
                ah1.q[1] = *(lds_cu32x4*)(cur + aRow1 + kk + aK0 + 16);
                const int ko = kk * 16;          // (kk/32)*FRAG_ELEMS
#pragma unroll
                for (int c = 0; c < 4; ++c) {
                    Frag bi, bh;
                    bi.q[0] = *(const u32x4*)(pi[c] + ko);
                    bi.q[1] = *(const u32x4*)(pi[c] + ko + 8);
                    bh.q[0] = *(const u32x4*)(ph[c] + ko);
                    bh.q[1] = *(const u32x4*)(ph[c] + ko + 8);
                    acc[0][c] = __builtin_amdgcn_wmma_f32_16x16x32_bf16(
                        false, ax0.f, false, bi.f, (short)0, acc[0][c], false, false);
                    acc[1][c] = __builtin_amdgcn_wmma_f32_16x16x32_bf16(
                        false, ax1.f, false, bi.f, (short)0, acc[1][c], false, false);
                    acc[0][c] = __builtin_amdgcn_wmma_f32_16x16x32_bf16(
                        false, ah0.f, false, bh.f, (short)0, acc[0][c], false, false);
                    acc[1][c] = __builtin_amdgcn_wmma_f32_16x16x32_bf16(
                        false, ah1.f, false, bh.f, (short)0, acc[1][c], false, false);
                }
            }
            // ---- epilogue: native v_tanh -> next-h (bf16 LDS) or output (f32) ----
            if (k < KSTEPS - 1) {
#pragma unroll
                for (int tr = 0; tr < 2; ++tr)
#pragma unroll
                    for (int c = 0; c < 4; ++c) {
                        const int col = nb + c * 16 + l16;
#pragma unroll
                        for (int v = 0; v < 8; ++v) {
                            const int row = tr * 16 + v + half * 8;   // C/D: M = v + 8*half
                            nxt[row * LDP + col] = f2bf(tanh_fast(acc[tr][c][v]));
                        }
                    }
            } else {
#pragma unroll
                for (int tr = 0; tr < 2; ++tr)
#pragma unroll
                    for (int c = 0; c < 4; ++c) {
                        const int col = nb + c * 16 + l16;
#pragma unroll
                        for (int v = 0; v < 8; ++v) {
                            const int row = tr * 16 + v + half * 8;
                            out[(size_t)(r0 + row) * DDIM + col] = tanh_fast(acc[tr][c][v]);
                        }
                    }
            }
        }
    }
}

extern "C" void kernel_launch(void* const* d_in, const int* in_sizes, int n_in,
                              void* d_out, int out_size, void* d_ws, size_t ws_size,
                              hipStream_t stream) {
    (void)in_sizes; (void)n_in; (void)out_size; (void)ws_size;
    const float* x   = (const float*)d_in[0];
    const float* Wih = (const float*)d_in[1];
    const float* Whh = (const float*)d_in[2];
    const float* bih = (const float*)d_in[3];
    const float* bhh = (const float*)d_in[4];
    float* out = (float*)d_out;

    bf16* wpi   = (bf16*)d_ws;                       //  2 MB packed W_ih
    bf16* wph   = wpi + DDIM * DDIM;                 //  2 MB packed W_hh
    float* bias = (float*)(wph + DDIM * DDIM);       //  4 KB
    bf16* xbf   = (bf16*)(bias + DDIM);              // 64 MB bf16 x

    pack_kernel<<<256, 256, 0, stream>>>(Wih, Whh, bih, bhh, wpi, wph, bias);
    xconv_kernel<<<(NROWS * KSTEPS * DDIM) / (256 * 8), 256, 0, stream>>>(x, xbf);

    hipFuncSetAttribute((const void*)rnnpool_kernel,
                        hipFuncAttributeMaxDynamicSharedMemorySize, SMEM_BYTES);
    rnnpool_kernel<<<NROWS / MTILE, 256, SMEM_BYTES, stream>>>(xbf, wpi, wph, bias, out);
}
